// AbilityEncoder_7954279432440
// MI455X (gfx1250) — matrix operand order repaired and checked by reading, not compiled, not run on hardware
//
#include <hip/hip_runtime.h>

// Fused AbilityEncoder for MI455X (gfx1250, wave32, WMMA bf16).
// One workgroup (8 waves / 256 threads) processes 16 samples end-to-end:
//   level2: 1 wave = 1 sample tile (M=16 nodes), x2 samples per wave
//   level1: waves 0-3, each a 16-row tile = 4 samples x 4 nodes
//   level0: wave 0, 16-row tile = 16 samples
// Weights live in LDS as bf16 (transposed for B-fragment loads); trig/eff/targ
// tables live in LDS as f32 (prim summed in f32); op_table lives in LDS as
// bf16 (only ever used as a WMMA A operand). All GEMMs use
// v_wmma_f32_16x16x32_bf16 with f32 accumulation.

typedef __bf16 bf16;
typedef __attribute__((ext_vector_type(16))) __bf16 v16bf;
typedef __attribute__((ext_vector_type(8)))  __bf16 v8bf;
typedef __attribute__((ext_vector_type(4)))  __bf16 v4bf;
typedef __attribute__((ext_vector_type(8)))  float  v8f;
typedef __attribute__((ext_vector_type(4)))  float  v4f;

#define WMMA_BF16(a, b, c) \
  __builtin_amdgcn_wmma_f32_16x16x32_bf16(false, (a), false, (b), (short)0, (c), false, false)

// ---------------- LDS layout (bytes) ----------------
#define OFF_TRIG 0                 // f32 [7][96]
#define OFF_EFF  2688              // f32 [9][96]
#define OFF_TARG 6144              // f32 [10][96]
#define OFF_OPB  9984              // bf16 [15][96]
#define OFF_B1   12864             // f32 [96]
#define OFF_B2   13248
#define OFF_BL   13632
#define OFF_W1T  14016             // bf16 [96][192]
#define OFF_W2T  50880             // bf16 [96][96]
#define OFF_WLT  69312             // bf16 [96][96]
#define OFF_HID  87744             // bf16, 8 waves x [16][96]
#define OFF_P2   112320            // bf16, 8 waves x [16][96] (pooled2 scratch)
#define OFF_P1   136896            // bf16 [16 samples][4][96] (pooled1)
#define OFF_P0   149184            // bf16 [16 samples][96]    (pooled0)
#define SMEM_BYTES 152256

// A fragment (16x32 bf16): lane holds row m = lane&15; chunks are
// cols [cb + hf*8 .. +7] and [cb + 16 + hf*8 .. +7]  (hf = lane>>4)
__device__ __forceinline__ v16bf ldsAFrag(const bf16* rows, int m, int stride,
                                          int cb, int hf) {
  const bf16* p0 = rows + m * stride + cb + hf * 8;
  v8bf lo = *(const v8bf*)p0;         // 16B ds_load_b128
  v8bf hi = *(const v8bf*)(p0 + 16);  // 16B ds_load_b128
  return __builtin_shufflevector(lo, hi, 0, 1, 2, 3, 4, 5, 6, 7,
                                 8, 9, 10, 11, 12, 13, 14, 15);
}

// A fragment when the lane already holds its own row pointer (op embedding).
__device__ __forceinline__ v16bf rowAFrag(const bf16* row, int cb, int hf) {
  const bf16* p0 = row + cb + hf * 8;
  v8bf lo = *(const v8bf*)p0;
  v8bf hi = *(const v8bf*)(p0 + 16);
  return __builtin_shufflevector(lo, hi, 0, 1, 2, 3, 4, 5, 6, 7,
                                 8, 9, 10, 11, 12, 13, 14, 15);
}

// B fragment (32x16 bf16) from W^T (row-major [N][K]): lane holds column
// n = lane&15; K chunks [kb + hf*8 .. +7] and [kb + 16 + hf*8 .. +7].
__device__ __forceinline__ v16bf ldsBFrag(const bf16* wt, int col, int stride,
                                          int kb, int hf) {
  const bf16* p = wt + col * stride + kb + hf * 8;
  v8bf lo = *(const v8bf*)p;
  v8bf hi = *(const v8bf*)(p + 16);
  return __builtin_shufflevector(lo, hi, 0, 1, 2, 3, 4, 5, 6, 7,
                                 8, 9, 10, 11, 12, 13, 14, 15);
}

__device__ __forceinline__ v8f splat8(float v) {
  v8f r = {v, v, v, v, v, v, v, v};
  return r;
}

// One 16-row node-MLP tile:  out = relu([op|pooled] @ W1 + b1) @ W2 + b2,
// optionally blended with leaf = relu(pooled @ Wl + bl) per-row (leafBits).
template <bool DO_LEAF>
__device__ __forceinline__ void tree_mlp_tile(
    int lane, const bf16* pooledRows, const bf16* opRow, bf16* hid,
    const bf16* W1T, const bf16* W2T, const bf16* WlT,
    const float* b1s, const float* b2s, const float* bls,
    unsigned leafBits, v8f* out) {
  const int m = lane & 15;
  const int hf = lane >> 4;

  v16bf aop[3], apl[3];
#pragma unroll
  for (int kb = 0; kb < 3; ++kb) {
    aop[kb] = rowAFrag(opRow, kb * 32, hf);
    apl[kb] = ldsAFrag(pooledRows, m, 96, kb * 32, hf);
  }

  // GEMM1: (16x192)@(192x96) + b1, ReLU -> hidden (bf16 in LDS scratch)
#pragma unroll
  for (int nt = 0; nt < 6; ++nt) {
    const int col = nt * 16 + m;
    v8f h = splat8(b1s[col]);
#pragma unroll
    for (int kb = 0; kb < 3; ++kb)
      h = WMMA_BF16(aop[kb], ldsBFrag(W1T, col, 192, kb * 32, hf), h);
#pragma unroll
    for (int kb = 0; kb < 3; ++kb)
      h = WMMA_BF16(apl[kb], ldsBFrag(W1T, col, 192, 96 + kb * 32, hf), h);
#pragma unroll
    for (int r = 0; r < 8; ++r)
      hid[(r + 8 * hf) * 96 + col] = (bf16)fmaxf(h[r], 0.0f);
  }

  // GEMM2: hidden(16x96) @ W2(96x96) + b2  (+ optional leaf blend)
  v16bf ah[3];
#pragma unroll
  for (int kb = 0; kb < 3; ++kb) ah[kb] = ldsAFrag(hid, m, 96, kb * 32, hf);

#pragma unroll
  for (int nt = 0; nt < 6; ++nt) {
    const int col = nt * 16 + m;
    v8f c = splat8(b2s[col]);
#pragma unroll
    for (int kb = 0; kb < 3; ++kb)
      c = WMMA_BF16(ah[kb], ldsBFrag(W2T, col, 96, kb * 32, hf), c);
    if (DO_LEAF) {
      v8f l = splat8(bls[col]);
#pragma unroll
      for (int kb = 0; kb < 3; ++kb)
        l = WMMA_BF16(apl[kb], ldsBFrag(WlT, col, 96, kb * 32, hf), l);
#pragma unroll
      for (int r = 0; r < 8; ++r)
        c[r] = ((leafBits >> r) & 1u) ? fmaxf(l[r], 0.0f) : c[r];
    }
    out[nt] = c;
  }
}

__global__ __launch_bounds__(256) void ability_encoder_kernel(
    const int* __restrict__ trig_ids, const int* __restrict__ act_ids,
    const int* __restrict__ targ_ids, const int* __restrict__ op_ids,
    const float* __restrict__ trig_table, const float* __restrict__ eff_table,
    const float* __restrict__ targ_table, const float* __restrict__ op_table,
    const float* __restrict__ W1, const float* __restrict__ b1,
    const float* __restrict__ W2, const float* __restrict__ b2,
    const float* __restrict__ Wl, const float* __restrict__ bl,
    float* __restrict__ out) {
  extern __shared__ char smem[];
  float* sTrig = (float*)(smem + OFF_TRIG);
  float* sEff  = (float*)(smem + OFF_EFF);
  float* sTarg = (float*)(smem + OFF_TARG);
  bf16*  sOpB  = (bf16*)(smem + OFF_OPB);
  float* sB1   = (float*)(smem + OFF_B1);
  float* sB2   = (float*)(smem + OFF_B2);
  float* sBl   = (float*)(smem + OFF_BL);
  bf16*  sW1T  = (bf16*)(smem + OFF_W1T);
  bf16*  sW2T  = (bf16*)(smem + OFF_W2T);
  bf16*  sWlT  = (bf16*)(smem + OFF_WLT);
  bf16*  sHid  = (bf16*)(smem + OFF_HID);
  bf16*  sP2   = (bf16*)(smem + OFF_P2);
  bf16*  sP1   = (bf16*)(smem + OFF_P1);
  bf16*  sP0   = (bf16*)(smem + OFF_P0);

  const int tid  = threadIdx.x;
  const int wave = tid >> 5;
  const int lane = tid & 31;
  const int m    = lane & 15;
  const int hf   = lane >> 4;
  const int S0   = blockIdx.x * 16;

  // ---- cooperative preload ----
  for (int i = tid; i < 7 * 96; i += 256)  sTrig[i] = trig_table[i];
  for (int i = tid; i < 9 * 96; i += 256)  sEff[i]  = eff_table[i];
  for (int i = tid; i < 10 * 96; i += 256) sTarg[i] = targ_table[i];
  for (int i = tid; i < 15 * 96; i += 256) sOpB[i]  = (bf16)op_table[i];
  for (int i = tid; i < 96; i += 256) {
    sB1[i] = b1[i]; sB2[i] = b2[i]; sBl[i] = bl[i];
  }
  for (int i = tid; i < 192 * 96; i += 256) {
    int k = i / 96, c = i % 96;
    sW1T[c * 192 + k] = (bf16)W1[i];
  }
  for (int i = tid; i < 96 * 96; i += 256) {
    int k = i / 96, c = i % 96;
    sW2T[c * 96 + k] = (bf16)W2[i];
  }
  for (int i = tid; i < 96 * 96; i += 256) {
    int k = i / 96, c = i % 96;
    sWlT[c * 96 + k] = (bf16)Wl[i];
  }
  __syncthreads();

  bf16* hid = sHid + wave * 16 * 96;
  v8f acc[6];

  // ================= level 2: nodes 5..20, 1 wave = 1 sample, x2 =================
#pragma unroll 1
  for (int rep = 0; rep < 2; ++rep) {
    const int sl = wave * 2 + rep;
    const int s  = S0 + sl;
    const int* tI = trig_ids + s * 21;
    const int* aI = act_ids  + s * 21;
    const int* gI = targ_ids + s * 21;
    const int* oI = op_ids   + s * 21;

    // Build pooled2 rows (prim[5+m]) into this wave's scratch, bf16.
    bf16* prow = sP2 + wave * 16 * 96 + m * 96;
    {
      const float* tr = sTrig + tI[5 + m] * 96;
      const float* er = sEff  + aI[5 + m] * 96;
      const float* gr = sTarg + gI[5 + m] * 96;
#pragma unroll
      for (int c4 = 0; c4 < 12; ++c4) {
        const int c = hf * 48 + c4 * 4;
        v4f v0 = *(const v4f*)(tr + c);
        v4f v1 = *(const v4f*)(er + c);
        v4f v2 = *(const v4f*)(gr + c);
        v4bf o;
        o[0] = (bf16)(v0[0] + v1[0] + v2[0]);
        o[1] = (bf16)(v0[1] + v1[1] + v2[1]);
        o[2] = (bf16)(v0[2] + v1[2] + v2[2]);
        o[3] = (bf16)(v0[3] + v1[3] + v2[3]);
        *(v4bf*)(prow + c) = o;  // 8B ds_store_b64
      }
    }
    // Leaf mask: bit r <-> D-row (r + 8*hf) <-> node 5 + r + 8*hf
    unsigned leafBits = 0;
#pragma unroll
    for (int r = 0; r < 8; ++r)
      if (oI[5 + r + 8 * hf] == 0) leafBits |= (1u << r);

    const bf16* opRow = sOpB + oI[5 + m] * 96;
    tree_mlp_tile<true>(lane, sP2 + wave * 16 * 96, opRow, hid,
                        sW1T, sW2T, sWlT, sB1, sB2, sBl, leafBits, acc);

    // child1[j] = sum of 4 consecutive rows; write pooled1 = prim[1+j] + child1
    const int j0 = 2 * hf, j1 = 2 * hf + 1;
    const int t0 = tI[1 + j0], a0 = aI[1 + j0], g0 = gI[1 + j0];
    const int t1 = tI[1 + j1], a1 = aI[1 + j1], g1 = gI[1 + j1];
#pragma unroll
    for (int nt = 0; nt < 6; ++nt) {
      const int col = nt * 16 + m;
      v8f o = acc[nt];
      float s03 = (o[0] + o[1]) + (o[2] + o[3]);
      float s47 = (o[4] + o[5]) + (o[6] + o[7]);
      float p0v = sTrig[t0 * 96 + col] + sEff[a0 * 96 + col] + sTarg[g0 * 96 + col];
      float p1v = sTrig[t1 * 96 + col] + sEff[a1 * 96 + col] + sTarg[g1 * 96 + col];
      sP1[(sl * 4 + j0) * 96 + col] = (bf16)(p0v + s03);
      sP1[(sl * 4 + j1) * 96 + col] = (bf16)(p1v + s47);
    }
  }
  __syncthreads();

  // ================= level 1: nodes 1..4, waves 0..3, tile = 4 samples =================
  if (wave < 4) {
    const int t    = wave;
    const int ls   = t * 4 + (m >> 2);
    const int node = 1 + (m & 3);
    const int s    = S0 + ls;
    const bf16* opRow = sOpB + op_ids[s * 21 + node] * 96;
    tree_mlp_tile<false>(lane, sP1 + t * 16 * 96, opRow, hid,
                         sW1T, sW2T, sWlT, sB1, sB2, sBl, 0u, acc);

    const int j0 = 2 * hf, j1 = 2 * hf + 1;
    const int s0g = S0 + t * 4 + j0, s1g = S0 + t * 4 + j1;
    const int t0 = trig_ids[s0g * 21], a0 = act_ids[s0g * 21], g0 = targ_ids[s0g * 21];
    const int t1 = trig_ids[s1g * 21], a1 = act_ids[s1g * 21], g1 = targ_ids[s1g * 21];
#pragma unroll
    for (int nt = 0; nt < 6; ++nt) {
      const int col = nt * 16 + m;
      v8f o = acc[nt];
      float s03 = (o[0] + o[1]) + (o[2] + o[3]);
      float s47 = (o[4] + o[5]) + (o[6] + o[7]);
      float p0v = sTrig[t0 * 96 + col] + sEff[a0 * 96 + col] + sTarg[g0 * 96 + col];
      float p1v = sTrig[t1 * 96 + col] + sEff[a1 * 96 + col] + sTarg[g1 * 96 + col];
      sP0[(t * 4 + j0) * 96 + col] = (bf16)(p0v + s03);
      sP0[(t * 4 + j1) * 96 + col] = (bf16)(p1v + s47);
    }
  }
  __syncthreads();

  // ================= level 0: node 0, wave 0, tile = 16 samples =================
  if (wave == 0) {
    const int s = S0 + m;
    const bf16* opRow = sOpB + op_ids[s * 21] * 96;
    tree_mlp_tile<false>(lane, sP0, opRow, hid,
                         sW1T, sW2T, sWlT, sB1, sB2, sBl, 0u, acc);
    // D row (r + 8*hf) = sample S0 + r + 8*hf; lanes 0..15 of each half write
    // 16 consecutive f32 -> coalesced global stores.
#pragma unroll
    for (int nt = 0; nt < 6; ++nt) {
#pragma unroll
      for (int r = 0; r < 8; ++r) {
        out[(size_t)(S0 + r + 8 * hf) * 96 + nt * 16 + m] = acc[nt][r];
      }
    }
  }
}

extern "C" void kernel_launch(void* const* d_in, const int* in_sizes, int n_in,
                              void* d_out, int out_size, void* d_ws, size_t ws_size,
                              hipStream_t stream) {
  const int* trig_ids = (const int*)d_in[0];
  const int* act_ids  = (const int*)d_in[1];
  const int* targ_ids = (const int*)d_in[2];
  const int* op_ids   = (const int*)d_in[3];
  const float* trig_table = (const float*)d_in[4];
  const float* eff_table  = (const float*)d_in[5];
  const float* targ_table = (const float*)d_in[6];
  const float* op_table   = (const float*)d_in[7];
  const float* W1 = (const float*)d_in[8];
  const float* b1 = (const float*)d_in[9];
  const float* W2 = (const float*)d_in[10];
  const float* b2 = (const float*)d_in[11];
  const float* Wl = (const float*)d_in[12];
  const float* bl = (const float*)d_in[13];
  float* out = (float*)d_out;

  const int Nsamp = in_sizes[0] / 21;   // 32768
  const int ngrp  = Nsamp / 16;         // 2048 workgroups (N divisible by 16)

  ability_encoder_kernel<<<dim3(ngrp), dim3(256), SMEM_BYTES, stream>>>(
      trig_ids, act_ids, targ_ids, op_ids, trig_table, eff_table, targ_table,
      op_table, W1, b1, W2, b2, Wl, bl, out);
}
